// ArmLikenessGNN_43903155700065
// MI455X (gfx1250) — compile-verified, matrix operand order checked by reading, not run on hardware
//
#include <hip/hip_runtime.h>
#include <hip/hip_bf16.h>
#include <math.h>

#define HID 128
#define NG  64

typedef __attribute__((ext_vector_type(16))) __bf16 v16bf;
typedef __attribute__((ext_vector_type(8)))  float  v8f;

// ---------------------------------------------------------------- utilities
__global__ void fill_f32_kernel(float* p, float v, long cnt) {
    long t = (long)blockIdx.x * blockDim.x + threadIdx.x;
    if (t < cnt) p[t] = v;
}

__global__ void cvt_bf16_kernel(const float* __restrict__ s, __bf16* __restrict__ d, long cnt) {
    long t = (long)blockIdx.x * blockDim.x + threadIdx.x;
    if (t < cnt) d[t] = (__bf16)s[t];
}

__device__ __forceinline__ void atomicMaxF(float* a, float v) {
    if (v >= 0.0f) atomicMax((int*)a, __float_as_int(v));
    else           atomicMin((unsigned int*)a, __float_as_uint(v));
}

// ---------------------------------------------------------------- embedding
// axis/org MLP (3->16 relu LN -> 16), concat with "others" (13), nin 45->128
__global__ void embed_kernel(const float* __restrict__ x,
                             const float* __restrict__ aW1, const float* __restrict__ ab1,
                             const float* __restrict__ ag,  const float* __restrict__ abe,
                             const float* __restrict__ aW2, const float* __restrict__ ab2,
                             const float* __restrict__ oW1, const float* __restrict__ ob1,
                             const float* __restrict__ og,  const float* __restrict__ obe,
                             const float* __restrict__ oW2, const float* __restrict__ ob2,
                             const float* __restrict__ ninW, const float* __restrict__ ninb,
                             float* __restrict__ h0, int n)
{
    int node = blockIdx.x;
    if (node >= n) return;
    __shared__ float feat[45];
    __shared__ float tmp[32];
    __shared__ float zz[32];
    const int t = threadIdx.x;  // 128
    const float* xr = x + (size_t)node * 19;

    if (t < 32) {
        const int  i    = t & 15;
        const bool isOrg = (t >= 16);
        const float* in3 = xr + (isOrg ? 12 : 9);
        const float* W1  = isOrg ? oW1 : aW1;
        const float* b1  = isOrg ? ob1 : ab1;
        float s = b1[i];
        for (int k = 0; k < 3; ++k) s += in3[k] * W1[i * 3 + k];
        tmp[t] = fmaxf(s, 0.0f);
    }
    if (t >= 32 && t < 45) {
        int j = t - 32;
        feat[j] = (j < 9) ? xr[j] : xr[15 + (j - 9)];
    }
    __syncthreads();
    if (t < 32) {
        const int  i    = t & 15;
        const bool isOrg = (t >= 16);
        const int  base = isOrg ? 16 : 0;
        const float* g  = isOrg ? og  : ag;
        const float* be = isOrg ? obe : abe;
        float mu = 0.f;
        for (int k = 0; k < 16; ++k) mu += tmp[base + k];
        mu *= (1.0f / 16.0f);
        float var = 0.f;
        for (int k = 0; k < 16; ++k) { float d = tmp[base + k] - mu; var += d * d; }
        var *= (1.0f / 16.0f);
        float rs = rsqrtf(var + 1e-5f);
        zz[t] = (tmp[t] - mu) * rs * g[i] + be[i];
    }
    __syncthreads();
    if (t < 32) {
        const int  j    = t & 15;
        const bool isOrg = (t >= 16);
        const int  base = isOrg ? 16 : 0;
        const float* W2 = isOrg ? oW2 : aW2;
        const float* b2 = isOrg ? ob2 : ab2;
        float s = b2[j];
        for (int i = 0; i < 16; ++i) s += zz[base + i] * W2[j * 16 + i];
        feat[13 + base + j] = s;
    }
    __syncthreads();
    float o = ninb[t];
    for (int k = 0; k < 45; ++k) o += feat[k] * ninW[t * 45 + k];
    h0[(size_t)node * HID + t] = o;
}

// ---------------------------------------------------------------- edge scatter
// 32 lanes per edge, 4 channels each (float4 load, 4 fp32 global atomics)
__global__ void agg_kernel(const float* __restrict__ h,
                           const int* __restrict__ src, const int* __restrict__ dst,
                           float* __restrict__ agg, int nE)
{
    long t = (long)blockIdx.x * blockDim.x + threadIdx.x;
    int e = (int)(t >> 5);
    if (e >= nE) return;
    int cg = ((int)t & 31) * 4;
    int s = src[e], d = dst[e];
    const float4 v = *(const float4*)(h + (size_t)s * HID + cg);
    float* p = agg + (size_t)d * HID + cg;
    atomicAdd(p + 0, v.x); atomicAdd(p + 1, v.y);
    atomicAdd(p + 2, v.z); atomicAdd(p + 3, v.w);
}

// ---------------------------------------------------------------- WMMA helper
// One wave computes relu( A[16x128] @ W^T + bias ) -> Out[16][NCOLS]
// A lives in LDS row-major; fragments packed per CDNA5 16-bit layouts.
// Accumulators are zero-initialized (lets backend use inline SRC2=0);
// bias folded into the relu at the LDS write.
template <int NTILES, int OUTW>
__device__ __forceinline__ void wave_gemm_relu(const __bf16 (&A)[16][HID],
                                               const __bf16* __restrict__ W,
                                               const float* __restrict__ bias,
                                               int m16, int koff, int hi,
                                               float (&Out)[16][OUTW])
{
#pragma unroll
    for (int nt = 0; nt < NTILES; ++nt) {
        const int ncol = nt * 16 + m16;
        v8f c = {};
        const __bf16* wrow = W + (size_t)ncol * HID;  // B(k,n) = W[n][k]
#pragma unroll
        for (int ks = 0; ks < 4; ++ks) {
            v16bf a, b;
#pragma unroll
            for (int j = 0; j < 8; ++j) {
                const int base = (j < 4) ? (koff + 2 * j) : (16 + koff + 2 * (j - 4));
                const int k = ks * 32 + base;
                a[2 * j]     = A[m16][k];
                a[2 * j + 1] = A[m16][k + 1];
                b[2 * j]     = wrow[k];
                b[2 * j + 1] = wrow[k + 1];
            }
            c = __builtin_amdgcn_wmma_f32_16x16x32_bf16(false, a, false, b,
                                                        (short)0, c, false, false);
        }
        const float bv = bias[ncol];
#pragma unroll
        for (int r = 0; r < 8; ++r)
            Out[r + hi][ncol] = fmaxf(c[r] + bv, 0.0f);
    }
}

// ---------------------------------------------------------------- fused conv layer
// m=(1+eps)h+agg; t=LN(relu(m@W1^T+b1))*g1+be1; u=t@W2^T+b2; h'=LN(relu(u))*lng+lnb + h
__launch_bounds__(128)
__global__ void conv_kernel(const float* __restrict__ h, const float* __restrict__ agg,
                            const __bf16* __restrict__ W1, const float* __restrict__ b1,
                            const float* __restrict__ g1,  const float* __restrict__ be1,
                            const __bf16* __restrict__ W2, const float* __restrict__ b2,
                            const float* __restrict__ lng, const float* __restrict__ lnb,
                            const float* __restrict__ epsArr, int layer,
                            float* __restrict__ hout, int n)
{
    __shared__ __bf16 As[4][16][HID];   // 16 KB
    __shared__ float  Outs[4][16][HID]; // 32 KB
    const int wave = threadIdx.x >> 5;
    const int lane = threadIdx.x & 31;
    const int nodeBase = blockIdx.x * 64 + wave * 16;
    const float epsv = 1.0f + epsArr[layer];
    const int m16  = lane & 15;
    const int koff = (lane < 16) ? 0 : 8;
    const int hi   = (lane >> 4) << 3;

    // pull L2-resident weights toward the WGP ahead of first K-step
    __builtin_prefetch(W1 + (size_t)threadIdx.x * 128, 0, 1);
    __builtin_prefetch(W2 + (size_t)threadIdx.x * 128, 0, 1);

    // Phase 1: m -> bf16 tile
    for (int idx = lane; idx < 16 * HID; idx += 32) {
        int r = idx >> 7, c = idx & 127;
        int node = nodeBase + r;
        float v = 0.0f;
        if (node < n) {
            size_t o = (size_t)node * HID + c;
            v = epsv * h[o] + agg[o];
        }
        As[wave][r][c] = (__bf16)v;
    }
    __syncthreads();

    // Phase 2: t_pre = relu(m @ W1^T + b1)
    wave_gemm_relu<8, HID>(As[wave], W1, b1, m16, koff, hi, Outs[wave]);
    __syncthreads();

    // Phase 3: LN over channels per row -> bf16 tile (with g1/be1)
    for (int r = 0; r < 16; ++r) {
        float p[4];
#pragma unroll
        for (int q = 0; q < 4; ++q) p[q] = Outs[wave][r][lane * 4 + q];
        float s = p[0] + p[1] + p[2] + p[3];
        for (int off = 16; off >= 1; off >>= 1) s += __shfl_xor(s, off, 32);
        float mu = s * (1.0f / HID);
        float vs = 0.f;
#pragma unroll
        for (int q = 0; q < 4; ++q) { float d = p[q] - mu; vs += d * d; }
        for (int off = 16; off >= 1; off >>= 1) vs += __shfl_xor(vs, off, 32);
        float rs = rsqrtf(vs * (1.0f / HID) + 1e-5f);
#pragma unroll
        for (int q = 0; q < 4; ++q) {
            int c = lane * 4 + q;
            As[wave][r][c] = (__bf16)((p[q] - mu) * rs * g1[c] + be1[c]);
        }
    }
    __syncthreads();

    // Phase 4: u = relu(t @ W2^T + b2)   (relu is pre-final-LN per reference)
    wave_gemm_relu<8, HID>(As[wave], W2, b2, m16, koff, hi, Outs[wave]);
    __syncthreads();

    // Phase 5: h' = LN(relu(u))*lng+lnb + residual
    for (int r = 0; r < 16; ++r) {
        float p[4];
#pragma unroll
        for (int q = 0; q < 4; ++q) p[q] = Outs[wave][r][lane * 4 + q];
        float s = p[0] + p[1] + p[2] + p[3];
        for (int off = 16; off >= 1; off >>= 1) s += __shfl_xor(s, off, 32);
        float mu = s * (1.0f / HID);
        float vs = 0.f;
#pragma unroll
        for (int q = 0; q < 4; ++q) { float d = p[q] - mu; vs += d * d; }
        for (int off = 16; off >= 1; off >>= 1) vs += __shfl_xor(vs, off, 32);
        float rs = rsqrtf(vs * (1.0f / HID) + 1e-5f);
        int node = nodeBase + r;
        if (node < n) {
#pragma unroll
            for (int q = 0; q < 4; ++q) {
                int c = lane * 4 + q;
                hout[(size_t)node * HID + c] =
                    (p[q] - mu) * rs * lng[c] + lnb[c] + h[(size_t)node * HID + c];
            }
        }
    }
}

// ---------------------------------------------------------------- gate MLP (128->64->1), WMMA
// hidden = relu(h @ W1^T + b1); z = LN(hidden)*g+be; gate = z . W2 + b2
__launch_bounds__(128)
__global__ void gate_kernel(const float* __restrict__ h,
                            const __bf16* __restrict__ W1, const float* __restrict__ b1,
                            const float* __restrict__ g,  const float* __restrict__ be,
                            const float* __restrict__ W2, const float* __restrict__ b2,
                            float* __restrict__ gate, int n)
{
    __shared__ __bf16 As[4][16][HID];  // 16 KB
    __shared__ float  Outs[4][16][64]; // 16 KB
    const int wave = threadIdx.x >> 5;
    const int lane = threadIdx.x & 31;
    const int nodeBase = blockIdx.x * 64 + wave * 16;
    const int m16  = lane & 15;
    const int koff = (lane < 16) ? 0 : 8;
    const int hi   = (lane >> 4) << 3;

    __builtin_prefetch(W1 + (size_t)threadIdx.x * 64, 0, 1);

    // Phase 1: h -> bf16 tile
    for (int idx = lane; idx < 16 * HID; idx += 32) {
        int r = idx >> 7, c = idx & 127;
        int node = nodeBase + r;
        float v = (node < n) ? h[(size_t)node * HID + c] : 0.0f;
        As[wave][r][c] = (__bf16)v;
    }
    __syncthreads();

    // Phase 2: hidden = relu(h @ W1^T + b1), 64 outputs -> 4 N-tiles
    wave_gemm_relu<4, 64>(As[wave], W1, b1, m16, koff, hi, Outs[wave]);
    __syncthreads();

    // Phase 3: per-row LN over 64 + dot with W2 (2 channels per lane)
    const int c0 = lane * 2, c1 = c0 + 1;
    const float g0 = g[c0],  gv1 = g[c1];
    const float e0 = be[c0], e1 = be[c1];
    const float w0 = W2[c0], w1 = W2[c1];
    for (int r = 0; r < 16; ++r) {
        float p0 = Outs[wave][r][c0];
        float p1 = Outs[wave][r][c1];
        float s = p0 + p1;
        for (int off = 16; off >= 1; off >>= 1) s += __shfl_xor(s, off, 32);
        float mu = s * (1.0f / 64.0f);
        float d0 = p0 - mu, d1 = p1 - mu;
        float vs = d0 * d0 + d1 * d1;
        for (int off = 16; off >= 1; off >>= 1) vs += __shfl_xor(vs, off, 32);
        float rs = rsqrtf(vs * (1.0f / 64.0f) + 1e-5f);
        float z0 = d0 * rs * g0 + e0;
        float z1 = d1 * rs * gv1 + e1;
        float dot = z0 * w0 + z1 * w1;
        for (int off = 16; off >= 1; off >>= 1) dot += __shfl_xor(dot, off, 32);
        int node = nodeBase + r;
        if (lane == 0 && node < n) gate[node] = dot + b2[0];
    }
}

// ---------------------------------------------------------------- softmax pooling
__global__ void segmax_kernel(const float* __restrict__ gate, const int* __restrict__ batch,
                              float* __restrict__ gmax, int n)
{
    int t = blockIdx.x * blockDim.x + threadIdx.x;
    if (t >= n) return;
    atomicMaxF(&gmax[batch[t]], gate[t]);
}

__global__ void segexp_kernel(const float* __restrict__ gate, const int* __restrict__ batch,
                              const float* __restrict__ gmax, const float* __restrict__ h,
                              float* __restrict__ gnum, float* __restrict__ gden, int n)
{
    long t = (long)blockIdx.x * blockDim.x + threadIdx.x;
    if (t >= (long)n * HID) return;
    int node = (int)(t >> 7), c = (int)t & 127;
    int b = batch[node];
    float e = __expf(gate[node] - gmax[b]);
    atomicAdd(&gnum[b * HID + c], e * h[(size_t)node * HID + c]);
    if (c == 0) atomicAdd(&gden[b], e);
}

// ---------------------------------------------------------------- head MLP per graph
__global__ void head_kernel(const float* __restrict__ gnum, const float* __restrict__ gden,
                            const float* __restrict__ W1, const float* __restrict__ b1,
                            const float* __restrict__ g,  const float* __restrict__ be,
                            const float* __restrict__ W2, const float* __restrict__ b2,
                            float* __restrict__ out)
{
    int gi = blockIdx.x;
    __shared__ float gv[HID];
    __shared__ float hh[64];
    __shared__ float zz[64];
    const int t = threadIdx.x;  // 64
    float inv = 1.0f / gden[gi];
    gv[t]      = gnum[gi * HID + t] * inv;
    gv[t + 64] = gnum[gi * HID + 64 + t] * inv;
    __syncthreads();
    float s = b1[t];
    for (int k = 0; k < HID; ++k) s += gv[k] * W1[t * HID + k];
    hh[t] = fmaxf(s, 0.0f);
    __syncthreads();
    float mu = 0.f;
    for (int k = 0; k < 64; ++k) mu += hh[k];
    mu *= (1.0f / 64.0f);
    float var = 0.f;
    for (int k = 0; k < 64; ++k) { float d = hh[k] - mu; var += d * d; }
    var *= (1.0f / 64.0f);
    float rs = rsqrtf(var + 1e-5f);
    zz[t] = (hh[t] - mu) * rs * g[t] + be[t];
    __syncthreads();
    if (t == 0) {
        float o = b2[0];
        for (int k = 0; k < 64; ++k) o += zz[k] * W2[k];
        out[gi] = o;
    }
}

// ---------------------------------------------------------------- launcher
extern "C" void kernel_launch(void* const* d_in, const int* in_sizes, int n_in,
                              void* d_out, int out_size, void* d_ws, size_t ws_size,
                              hipStream_t stream)
{
    const float* x    = (const float*)d_in[0];
    const int*   ei   = (const int*)d_in[1];
    const int*   batch = (const int*)d_in[2];
    const float* aW1 = (const float*)d_in[3],  *ab1 = (const float*)d_in[4];
    const float* ag  = (const float*)d_in[5],  *abe = (const float*)d_in[6];
    const float* aW2 = (const float*)d_in[7],  *ab2 = (const float*)d_in[8];
    const float* oW1 = (const float*)d_in[9],  *ob1 = (const float*)d_in[10];
    const float* og  = (const float*)d_in[11], *obe = (const float*)d_in[12];
    const float* oW2 = (const float*)d_in[13], *ob2 = (const float*)d_in[14];
    const float* ninW = (const float*)d_in[15], *ninb = (const float*)d_in[16];
    const float* cW1 = (const float*)d_in[17], *cb1 = (const float*)d_in[18];
    const float* cg  = (const float*)d_in[19], *cbe = (const float*)d_in[20];
    const float* cW2 = (const float*)d_in[21], *cb2 = (const float*)d_in[22];
    const float* eps = (const float*)d_in[23];
    const float* lng = (const float*)d_in[24], *lnb = (const float*)d_in[25];
    const float* gW1 = (const float*)d_in[26], *gb1 = (const float*)d_in[27];
    const float* gg  = (const float*)d_in[28], *gbe = (const float*)d_in[29];
    const float* gW2 = (const float*)d_in[30], *gb2 = (const float*)d_in[31];
    const float* hW1 = (const float*)d_in[32], *hb1 = (const float*)d_in[33];
    const float* hg  = (const float*)d_in[34], *hbe = (const float*)d_in[35];
    const float* hW2 = (const float*)d_in[36], *hb2 = (const float*)d_in[37];

    const int n  = in_sizes[0] / 19;
    const int nE = in_sizes[1] / 2;
    const size_t hbytes = (size_t)n * HID * sizeof(float);
    auto al = [](size_t s) { return (s + 255) & ~(size_t)255; };

    char* w = (char*)d_ws;
    float*  hA   = (float*)w;  w += al(hbytes);
    float*  hB   = (float*)w;  w += al(hbytes);
    float*  agg  = (float*)w;  w += al(hbytes);
    float*  gate = (float*)w;  w += al((size_t)n * sizeof(float));
    float*  gmax = (float*)w;  w += al(NG * sizeof(float));
    float*  gden = (float*)w;  w += al(NG * sizeof(float));
    float*  gnum = (float*)w;  w += al(NG * HID * sizeof(float));
    __bf16* W1bf = (__bf16*)w; w += al((size_t)3 * HID * HID * sizeof(__bf16));
    __bf16* W2bf = (__bf16*)w; w += al((size_t)3 * HID * HID * sizeof(__bf16));
    __bf16* GWbf = (__bf16*)w; w += al((size_t)64 * HID * sizeof(__bf16));

    // bf16 weight conversion (deterministic, every call)
    {
        long cnt = 3L * HID * HID;
        int blk = 256, grd = (int)((cnt + blk - 1) / blk);
        cvt_bf16_kernel<<<grd, blk, 0, stream>>>(cW1, W1bf, cnt);
        cvt_bf16_kernel<<<grd, blk, 0, stream>>>(cW2, W2bf, cnt);
        cvt_bf16_kernel<<<(64 * HID + 255) / 256, 256, 0, stream>>>(gW1, GWbf, 64L * HID);
    }

    // input embedding -> hA
    embed_kernel<<<n, 128, 0, stream>>>(x, aW1, ab1, ag, abe, aW2, ab2,
                                        oW1, ob1, og, obe, oW2, ob2,
                                        ninW, ninb, hA, n);

    float* hc = hA; float* hn = hB;
    for (int l = 0; l < 3; ++l) {
        {
            long cnt = (long)n * HID;
            int blk = 256, grd = (int)((cnt + blk - 1) / blk);
            fill_f32_kernel<<<grd, blk, 0, stream>>>(agg, 0.0f, cnt);
        }
        {
            long work = (long)nE * 32;
            int blk = 256, grd = (int)((work + blk - 1) / blk);
            agg_kernel<<<grd, blk, 0, stream>>>(hc, ei, ei + nE, agg, nE);
        }
        {
            int grd = (n + 63) / 64;
            conv_kernel<<<grd, 128, 0, stream>>>(hc, agg,
                                                 W1bf + (size_t)l * HID * HID, cb1 + l * HID,
                                                 cg + l * HID, cbe + l * HID,
                                                 W2bf + (size_t)l * HID * HID, cb2 + l * HID,
                                                 lng + l * HID, lnb + l * HID,
                                                 eps, l, hn, n);
        }
        float* tmp = hc; hc = hn; hn = tmp;
    }

    // attention pooling
    gate_kernel<<<(n + 63) / 64, 128, 0, stream>>>(hc, GWbf, gb1, gg, gbe, gW2, gb2,
                                                   gate, n);
    fill_f32_kernel<<<1, NG, 0, stream>>>(gmax, -INFINITY, NG);
    fill_f32_kernel<<<1, NG, 0, stream>>>(gden, 0.0f, NG);
    fill_f32_kernel<<<(NG * HID + 255) / 256, 256, 0, stream>>>(gnum, 0.0f, NG * HID);
    segmax_kernel<<<(n + 255) / 256, 256, 0, stream>>>(gate, batch, gmax, n);
    {
        long cnt = (long)n * HID;
        int blk = 256, grd = (int)((cnt + blk - 1) / blk);
        segexp_kernel<<<grd, blk, 0, stream>>>(gate, batch, gmax, hc, gnum, gden, n);
    }
    head_kernel<<<NG, 64, 0, stream>>>(gnum, gden, hW1, hb1, hg, hbe, hW2, hb2,
                                       (float*)d_out);
}
